// Neighbor_SA_kaiming_20160576487494
// MI455X (gfx1250) — matrix operand order repaired and checked by reading, not compiled, not run on hardware
//
#include <hip/hip_runtime.h>
#include <hip/hip_bf16.h>
#include <math.h>

// ---------------------------------------------------------------------------
// Problem constants (from reference): B=8, N=1024, K=32, D=128, da=32
// ---------------------------------------------------------------------------
#define NB    8
#define NN    1024
#define NK    32
#define ND    128
#define NDA   32
#define NTILE (NB * NN)                 // 8192 independent (b,n) tiles
#define NELEM ((size_t)NTILE * NK * ND) // 33,554,432 f32 elements
#define EPS   1e-5f
#define INV_SQRT_DA 0.17677669529663687f

typedef __attribute__((ext_vector_type(16))) __bf16        v16bf;
typedef __attribute__((ext_vector_type(8)))  float         v8f;
typedef __attribute__((ext_vector_type(4)))  unsigned int  v4ui;
typedef __attribute__((ext_vector_type(4)))  int           v4i;
typedef __attribute__((ext_vector_type(8)))  int           v8i;

// ---------------------------------------------------------------------------
// WMMA helpers (CDNA5 wave32): D(16x16,f32) = A(16x32,bf16) * B(32x16,bf16) + C
// A-frag (16-bit, 16x32): lane m=L&15, half=L>>4; dword r<4 holds K=half*8+2r,
//   r>=4 holds K=16+half*8+2(r-4)  -> two contiguous 16B chunks per lane.
// B-frag (16-bit, 32x16): lane n=L&15, half=L>>4; same K order as A-frag
//   -> two contiguous 16B chunks per lane IF B is stored transposed [n][k].
// C/D (f32 16x16): lane n=L&15; vgpr r holds row m = (L>>4)*8 + r.
// ---------------------------------------------------------------------------
__device__ __forceinline__ v16bf load_a_frag(const __bf16* base, int ld, int lane) {
    int m = lane & 15, half = lane >> 4;
    union { v16bf v; uint4 u[2]; } r;
    r.u[0] = *reinterpret_cast<const uint4*>(base + m * ld + half * 8);
    r.u[1] = *reinterpret_cast<const uint4*>(base + m * ld + 16 + half * 8);
    return r.v;
}

__device__ __forceinline__ v16bf load_b_frag(const __bf16* base, int ld, int lane) {
    int n = lane & 15, half = lane >> 4;
    const __bf16* p = base + n * ld + half * 8;
    union { v16bf v; uint4 u[2]; } r;
    r.u[0] = *reinterpret_cast<const uint4*>(p);
    r.u[1] = *reinterpret_cast<const uint4*>(p + 16);
    return r.v;
}

__device__ __forceinline__ v8f wmma_bf16(v16bf a, v16bf b, v8f c) {
    // 8 args: (neg_a, A, neg_b, B, c_mod, C, reuse_a, reuse_b)
    return __builtin_amdgcn_wmma_f32_16x16x32_bf16(false, a, false, b, (short)0, c,
                                                   false, false);
}

__device__ __forceinline__ void store_d_bf16_rowmajor(__bf16* base, int ld, int lane,
                                                      v8f acc) {
    int n = lane & 15, half = lane >> 4;
#pragma unroll
    for (int r = 0; r < 8; ++r)
        base[(half * 8 + r) * ld + n] = (__bf16)acc[r];
}

// ---------------------------------------------------------------------------
// TDM: issue TENSOR_LOAD_TO_LDS for a 2D tile [rows x cols] of 4-byte elements
// with row stride `stride` (elements). One call per wave; tracked on TENSORcnt.
// D# layout per CDNA5 ISA.
// ---------------------------------------------------------------------------
__device__ __forceinline__ void tdm_load_tile_f32(unsigned int lds_byte_addr,
                                                  const float* gptr,
                                                  unsigned int rows,
                                                  unsigned int cols,
                                                  unsigned int stride) {
    unsigned long long ga = (unsigned long long)(uintptr_t)gptr;
    v4ui g0;
    g0.x = 1u;                                   // count=1, user mode, no gather
    g0.y = lds_byte_addr;                        // lds_addr [63:32]
    g0.z = (unsigned int)(ga & 0xffffffffu);     // global_addr [95:64]
    g0.w = (unsigned int)((ga >> 32) & 0x01ffffffu) | 0x80000000u; // addr[56:32] | type=2

    v8i g1;
    g1[0] = (int)(2u << 16);                     // wg_mask=0, data_size=2 (4B)
    g1[1] = (int)(cols << 16);                   // tensor_dim0[15:0] in [63:48]
    g1[2] = (int)((cols >> 16) | (rows << 16));  // tensor_dim0[31:16], tensor_dim1[15:0]
    g1[3] = (int)((rows >> 16) | (cols << 16));  // tensor_dim1[31:16], tile_dim0
    g1[4] = (int)rows;                           // tile_dim1 (tile_dim2=0)
    g1[5] = (int)stride;                         // tensor_dim0_stride[31:0]
    g1[6] = 0;                                   // stride[47:32], dim1_stride[15:0]
    g1[7] = 0;

    v4i g2 = {0, 0, 0, 0};                       // dims 2/3 unused (zeros)
    v4i g3 = {0, 0, 0, 0};

#if __has_include(<hip/amd_detail/amd_gfx1250_TDM.h>)
    // therock/clang-23 toolchain: 6-arg form (extra int32x8 group)
    v8i g4 = {0, 0, 0, 0, 0, 0, 0, 0};
    __builtin_amdgcn_tensor_load_to_lds(g0, g1, g2, g3, g4, 0);
#else
    // ROCm 7.2 / clang-22: 5-arg form
    __builtin_amdgcn_tensor_load_to_lds(g0, g1, g2, g3, 0);
#endif
}

// ---------------------------------------------------------------------------
// Kernel 0: zero BN accumulators; convert weights to bf16 in B-transposed
// [n][k] layout so WMMA B-fragments are contiguous loads.
//   wqT[e][d] = w_Q[d][e]   (4096)     wkT[e][d] = w_K[d][e]   (4096)
//   wvT[e][d] = w_V[d][e]   (16384)    cwb[c][d] = conv_w[c][d](16384, as-is)
// ---------------------------------------------------------------------------
__global__ __launch_bounds__(256) void prep_kernel(
    const float* __restrict__ wQ, const float* __restrict__ wK,
    const float* __restrict__ wV, const float* __restrict__ convw,
    __bf16* __restrict__ wqT, __bf16* __restrict__ wkT,
    __bf16* __restrict__ wvT, __bf16* __restrict__ cwb,
    float* __restrict__ gsum, float* __restrict__ gsq) {
    int tid = threadIdx.x;
    if (tid < ND) { gsum[tid] = 0.0f; gsq[tid] = 0.0f; }
    for (int i = tid; i < NDA * ND; i += 256) {
        int e = i / ND, d = i % ND;
        wqT[i] = (__bf16)wQ[d * NDA + e];
        wkT[i] = (__bf16)wK[d * NDA + e];
    }
    for (int i = tid; i < ND * ND; i += 256) {
        int e = i / ND, d = i % ND;
        wvT[i] = (__bf16)wV[d * ND + e];
        cwb[i] = (__bf16)convw[i];          // conv_w is [c][d] == B^T already
    }
}

// ---------------------------------------------------------------------------
// Kernel 1: fused attention + 1x1 conv, one workgroup (8 waves) per (b,n) tile.
// x tile is staged into LDS by the Tensor Data Mover (TENSOR_LOAD_TO_LDS),
// then converted to bf16 for the WMMA pipeline. Emits y (pre-BN, f32) and
// per-channel sum / sum^2 for BatchNorm.
// ---------------------------------------------------------------------------
__global__ __launch_bounds__(256) void fused_attn_kernel(
    const float* __restrict__ x,
    const __bf16* __restrict__ wqT, const __bf16* __restrict__ wkT,
    const __bf16* __restrict__ wvT, const __bf16* __restrict__ cwb,
    const float* __restrict__ conv_b,
    float* __restrict__ y_out, float* __restrict__ gsum, float* __restrict__ gsq) {

    __shared__ __align__(16) float  xstage[NK * ND];  // raw f32 tile via TDM (16KB)
    __shared__ __align__(16) __bf16 xt[NK * ND];      // x tile, bf16 [32][128]
    __shared__ __align__(16) __bf16 qs[NK * NDA];     // q [i][e]   [32][32]
    __shared__ __align__(16) __bf16 ks[NK * NDA];     // k [j][e]   [32][32]
    __shared__ __align__(16) __bf16 attA[NK * NK];    // A = att^T  [i][j]
    __shared__ __align__(16) __bf16 vT[ND * NK];      // v^T [d][j] [128][32]
    __shared__ __align__(16) __bf16 xs[NK * ND];      // x_s [i][d] [32][128]
    __shared__ __align__(16) float  es[NK * NK];      // energy f32 [32][32]
    __shared__ float csum[ND];
    __shared__ float csq[ND];

    const int tid  = threadIdx.x;
    const int wv   = tid >> 5;
    const int lane = tid & 31;
    const int bn   = blockIdx.x;                      // tile id in [0, 8192)
    const size_t tile_off = (size_t)bn * NK * ND;     // 4096 floats per tile

    // ---- TDM: DMA the 32x128 f32 tile into LDS (wave 0 issues, whole-wave op)
    if (wv == 0) {
        unsigned int lds_addr = (unsigned int)(uintptr_t)(&xstage[0]);
        tdm_load_tile_f32(lds_addr, x + tile_off, NK, ND, ND);
        __builtin_amdgcn_s_wait_tensorcnt(0);
    }
    if (tid < ND) { csum[tid] = 0.0f; csq[tid] = 0.0f; }
    __syncthreads();                                  // xstage visible to all waves

    // ---- convert staged tile to bf16 ----
    {
        const float4* xin = reinterpret_cast<const float4*>(xstage);
#pragma unroll
        for (int i = 0; i < 4; ++i) {
            int idx = tid + i * 256;                  // 1024 float4 per tile
            float4 f = xin[idx];
            __bf16* p = &xt[idx * 4];
            p[0] = (__bf16)f.x; p[1] = (__bf16)f.y;
            p[2] = (__bf16)f.z; p[3] = (__bf16)f.w;
        }
    }
    __syncthreads();

    // ---- phase A: q, k (2x2 tiles each) and v (2x8 tiles); 3 tiles / wave ----
#pragma unroll
    for (int s = 0; s < 3; ++s) {
        int id = wv * 3 + s;                          // 0..23
        const __bf16* wt;
        int mi, ni;
        bool isV = false;
        if (id < 4)       { wt = wqT; mi = id >> 1;        ni = id & 1; }
        else if (id < 8)  { wt = wkT; mi = (id - 4) >> 1;  ni = (id - 4) & 1; }
        else              { wt = wvT; mi = (id - 8) >> 3;  ni = (id - 8) & 7; isV = true; }

        v8f acc = {};
#pragma unroll
        for (int kk = 0; kk < 4; ++kk) {              // reduce over D=128
            v16bf a = load_a_frag(&xt[(mi * 16) * ND + kk * 32], ND, lane);
            v16bf b = load_b_frag(wt + (ni * 16) * ND + kk * 32, ND, lane);
            acc = wmma_bf16(a, b, acc);
        }
        if (!isV) {
            __bf16* dst = (id < 4) ? qs : ks;
            store_d_bf16_rowmajor(dst + (mi * 16) * NDA + ni * 16, NDA, lane, acc);
        } else {
            // transposed store: v[j][d] -> vT[d][j]; 8 rows contiguous = 16B
            int n = lane & 15, half = lane >> 4;
            union { __bf16 h[8]; uint4 u; } pk;
#pragma unroll
            for (int r = 0; r < 8; ++r) pk.h[r] = (__bf16)acc[r];
            *reinterpret_cast<uint4*>(&vT[(ni * 16 + n) * NK + mi * 16 + half * 8]) = pk.u;
        }
    }
    __syncthreads();

    // ---- phase B: energy = q @ k^T / sqrt(da); 4 tiles on waves 0..3 ----
    if (wv < 4) {
        int mi = wv >> 1, ni = wv & 1;
        v16bf a = load_a_frag(&qs[(mi * 16) * NDA], NDA, lane);
        v16bf b = load_b_frag(&ks[(ni * 16) * NDA], NDA, lane);  // ks is [n][k]
        v8f acc = {};
        acc = wmma_bf16(a, b, acc);
        int n = lane & 15, half = lane >> 4;
#pragma unroll
        for (int r = 0; r < 8; ++r)
            es[(mi * 16 + half * 8 + r) * NK + ni * 16 + n] = acc[r] * INV_SQRT_DA;
    }
    __syncthreads();

    // ---- phase C: softmax row i over j; write A = att^T : attA[j][i] ----
    if (tid < NK) {
        int i = tid;
        float m = -3.402823e38f;
#pragma unroll
        for (int j = 0; j < NK; ++j) m = fmaxf(m, es[i * NK + j]);
        float ssum = 0.0f;
        float e[NK];
#pragma unroll
        for (int j = 0; j < NK; ++j) { e[j] = __expf(es[i * NK + j] - m); ssum += e[j]; }
        float inv = 1.0f / ssum;
#pragma unroll
        for (int j = 0; j < NK; ++j)
            attA[j * NK + i] = (__bf16)(e[j] * inv);  // att[i][j] -> A[j][i]
    }
    __syncthreads();

    // ---- phase D: x_s = att^T @ v : A=attA [32][32], B=vT [d][j]; 2 tiles/wave ----
#pragma unroll
    for (int s = 0; s < 2; ++s) {
        int id = wv * 2 + s;                          // 0..15, 2x8 tiles
        int mi = id >> 3, ni = id & 7;
        v16bf a = load_a_frag(&attA[(mi * 16) * NK], NK, lane);
        v16bf b = load_b_frag(&vT[(ni * 16) * NK], NK, lane);
        v8f acc = {};
        acc = wmma_bf16(a, b, acc);
        store_d_bf16_rowmajor(xs + (mi * 16) * ND + ni * 16, ND, lane, acc);
    }
    __syncthreads();

    // ---- phase E: y = x_s @ conv_w^T + b; store f32 + channel stats ----
#pragma unroll
    for (int s = 0; s < 2; ++s) {
        int id = wv * 2 + s;
        int mi = id >> 3, ni = id & 7;
        v8f acc = {};
#pragma unroll
        for (int kk = 0; kk < 4; ++kk) {
            v16bf a = load_a_frag(&xs[(mi * 16) * ND + kk * 32], ND, lane);
            v16bf b = load_b_frag(cwb + (ni * 16) * ND + kk * 32, ND, lane);
            acc = wmma_bf16(a, b, acc);
        }
        int n = lane & 15, half = lane >> 4;
        int c = ni * 16 + n;
        float bias = conv_b[c];
        float lsum = 0.0f, lsq = 0.0f;
#pragma unroll
        for (int r = 0; r < 8; ++r) {
            float yv = acc[r] + bias;
            int m = mi * 16 + half * 8 + r;
            y_out[tile_off + (size_t)m * ND + c] = yv;
            lsum += yv;
            lsq  += yv * yv;
        }
        atomicAdd(&csum[c], lsum);
        atomicAdd(&csq[c],  lsq);
    }
    __syncthreads();

    if (tid < ND) {
        atomicAdd(&gsum[tid], csum[tid]);
        atomicAdd(&gsq[tid],  csq[tid]);
    }
}

// ---------------------------------------------------------------------------
// Kernel 2: per-channel BN scale/shift from accumulated sums.
// ---------------------------------------------------------------------------
__global__ void stats_kernel(const float* __restrict__ gsum,
                             const float* __restrict__ gsq,
                             const float* __restrict__ gamma,
                             const float* __restrict__ beta,
                             float* __restrict__ scale,
                             float* __restrict__ shift) {
    int c = threadIdx.x;
    if (c >= ND) return;
    const float M = (float)((size_t)NB * NN * NK);    // 262144
    float mean = gsum[c] / M;
    float var  = gsq[c] / M - mean * mean;
    float sc   = gamma[c] * rsqrtf(var + EPS);
    scale[c] = sc;
    shift[c] = beta[c] - mean * sc;
}

// ---------------------------------------------------------------------------
// Kernel 3: out = x + relu(y * scale[c] + shift[c])  (pure bandwidth, float4)
// ---------------------------------------------------------------------------
__global__ __launch_bounds__(256) void finalize_kernel(
    const float* __restrict__ x, const float* __restrict__ y,
    const float* __restrict__ scale, const float* __restrict__ shift,
    float* __restrict__ out) {
    size_t t = (size_t)blockIdx.x * 256 + threadIdx.x; // one float4 per thread
    float4 xv = reinterpret_cast<const float4*>(x)[t];
    float4 yv = reinterpret_cast<const float4*>(y)[t];
    int c = (int)((t * 4) & (ND - 1));                 // D divides row, no wrap
    float4 o;
    o.x = xv.x + fmaxf(0.0f, yv.x * scale[c + 0] + shift[c + 0]);
    o.y = xv.y + fmaxf(0.0f, yv.y * scale[c + 1] + shift[c + 1]);
    o.z = xv.z + fmaxf(0.0f, yv.z * scale[c + 2] + shift[c + 2]);
    o.w = xv.w + fmaxf(0.0f, yv.w * scale[c + 3] + shift[c + 3]);
    reinterpret_cast<float4*>(out)[t] = o;
}

// ---------------------------------------------------------------------------
// Host-side launch. Inputs (setup_inputs order):
//   0:x 1:w_Q 2:w_K 3:w_V 4:conv_w 5:conv_b 6:gamma 7:beta
// Workspace layout: [y f32 128MB][gsum 128][gsq 128][scale 128][shift 128]
//                   [wqT 4096 bf16][wkT 4096][wvT 16384][cwb 16384]
// ---------------------------------------------------------------------------
extern "C" void kernel_launch(void* const* d_in, const int* in_sizes, int n_in,
                              void* d_out, int out_size, void* d_ws, size_t ws_size,
                              hipStream_t stream) {
    const float* x      = (const float*)d_in[0];
    const float* wQ     = (const float*)d_in[1];
    const float* wK     = (const float*)d_in[2];
    const float* wV     = (const float*)d_in[3];
    const float* convw  = (const float*)d_in[4];
    const float* conv_b = (const float*)d_in[5];
    const float* gamma  = (const float*)d_in[6];
    const float* beta   = (const float*)d_in[7];
    float* out = (float*)d_out;

    char* ws = (char*)d_ws;
    const size_t Y_BYTES = NELEM * sizeof(float);     // 134,217,728
    float* y     = (float*)ws;
    float* gsum  = (float*)(ws + Y_BYTES);
    float* gsq   = gsum + ND;
    float* scale = gsum + 2 * ND;
    float* shift = gsum + 3 * ND;
    __bf16* wqT  = (__bf16*)(ws + Y_BYTES + 4 * ND * sizeof(float));
    __bf16* wkT  = wqT + NDA * ND;
    __bf16* wvT  = wkT + NDA * ND;
    __bf16* cwb  = wvT + ND * ND;

    prep_kernel<<<1, 256, 0, stream>>>(wQ, wK, wV, convw, wqT, wkT, wvT, cwb,
                                       gsum, gsq);
    fused_attn_kernel<<<NTILE, 256, 0, stream>>>(x, wqT, wkT, wvT, cwb, conv_b,
                                                 y, gsum, gsq);
    stats_kernel<<<1, 128, 0, stream>>>(gsum, gsq, gamma, beta, scale, shift);
    finalize_kernel<<<(unsigned)(NELEM / 4 / 256), 256, 0, stream>>>(x, y, scale,
                                                                     shift, out);
}
